// MPO3_12627203850775
// MI455X (gfx1250) — matrix operand order repaired and checked by reading, not compiled, not run on hardware
//
#include <hip/hip_runtime.h>
#include <hip/hip_bf16.h>

typedef __attribute__((ext_vector_type(16))) __bf16 v16bf;
typedef __attribute__((ext_vector_type(8)))  float  v8f;
typedef __attribute__((ext_vector_type(4)))  float  v4f;

union BfPack {
  unsigned short us[16];
  v4f            f4[2];
  v16bf          bf;
};

__device__ __forceinline__ unsigned short f2bf(float f) {
  unsigned int u = __float_as_uint(f);
  u += 0x7FFFu + ((u >> 16) & 1u);   // round-to-nearest-even
  return (unsigned short)(u >> 16);
}

// ---------------------------------------------------------------------------
// Kernel 1: materialize layer-0 map A[p,u] (1024x64) into WMMA B-operand
// packed bf16 tiles. Tile (kt,nt) = 32x16; lane L holds col n=L&15,
// rows k=(L>>4)*8+e for e<8, 16+(L>>4)*8+(e-8) for e>=8 (32B per lane).
// ---------------------------------------------------------------------------
__global__ __launch_bounds__(256) void build_A_packed(
    const float* __restrict__ t0, const float* __restrict__ t1,
    const float* __restrict__ t2, unsigned short* __restrict__ aPack) {
  int idx = blockIdx.x * 256 + threadIdx.x;   // 65536 threads
  int p = idx >> 6;                           // 0..1023
  int u = idx & 63;                           // 0..63
  int k0 = p >> 7, k1 = (p >> 4) & 7, k2 = p & 15;
  int m0 = u >> 4, m1 = (u >> 2) & 3, m2 = u & 3;
  float acc = 0.f;
#pragma unroll
  for (int a1 = 0; a1 < 4; ++a1) {
    float v0 = t0[a1 * 32 + k0 * 4 + m0];
#pragma unroll
    for (int a2 = 0; a2 < 4; ++a2)
      acc += v0 * t1[a1 * 128 + a2 * 32 + k1 * 4 + m1] *
                  t2[a2 * 64 + k2 * 4 + m2];
  }
  int kt = p >> 5, nt = u >> 4;
  int lane = (((p >> 3) & 1) << 4) | (u & 15);
  int e = (p & 7) | (((p >> 4) & 1) << 3);
  aPack[((kt * 4 + nt) * 32 + lane) * 16 + e] = f2bf(acc);
}

// ---------------------------------------------------------------------------
// Kernel 2: materialize layer-1 map W1[u,o] (64x32768) into packed bf16
// B-operand tiles: tile index kt*2048 + nt, kt in {0,1}, nt in [0,2048).
// ---------------------------------------------------------------------------
__global__ __launch_bounds__(256) void build_W1_packed(
    const float* __restrict__ t3, const float* __restrict__ t4,
    const float* __restrict__ t5, unsigned short* __restrict__ wPack) {
  int idx = blockIdx.x * 256 + threadIdx.x;   // 2097152 threads
  int u = idx >> 15;                          // 0..63
  int o = idx & 32767;                        // 0..32767
  int q0 = u >> 4, q1 = (u >> 2) & 3, q2 = u & 3;
  int m0 = o >> 10, m1 = (o >> 5) & 31, m2 = o & 31;
  float acc = 0.f;
#pragma unroll
  for (int a1 = 0; a1 < 4; ++a1) {
    float v0 = t3[a1 * 128 + q0 * 32 + m0];
#pragma unroll
    for (int a2 = 0; a2 < 4; ++a2)
      acc += v0 * t4[a1 * 512 + a2 * 128 + q1 * 32 + m1] *
                  t5[a2 * 128 + q2 * 32 + m2];
  }
  int kt = u >> 5, nt = o >> 4;
  int lane = (((u >> 3) & 1) << 4) | (o & 15);
  int e = (u & 7) | (((u >> 4) & 1) << 3);
  wPack[((kt * 2048 + nt) * 32 + lane) * 16 + e] = f2bf(acc);
}

// ---------------------------------------------------------------------------
// Kernel 3: h1 = x (4096x1024 f32) @ A  -> bf16 row-major (4096x64).
// One wave per (m_block, n_tile): 256 * 4 = 1024 waves. K = 1024 = 32 WMMAs.
// x converted to bf16 in-register (A-operand: lane holds row L&15,
// K = half*8 + {0..7} and 16 + half*8 + {0..7}).
// ---------------------------------------------------------------------------
__global__ __launch_bounds__(256) void gemm1(
    const float* __restrict__ x, const unsigned short* __restrict__ aPack,
    unsigned short* __restrict__ h1) {
  int wave = blockIdx.x * 8 + (threadIdx.x >> 5);  // 0..1023
  int lane = threadIdx.x & 31;
  int mb = wave >> 2;
  int nt = wave & 3;
  int half = lane >> 4;
  const float* xrow = x + (size_t)(mb * 16 + (lane & 15)) * 1024;
  v8f acc = {};
  for (int kt = 0; kt < 32; ++kt) {
    int cb = kt * 32 + half * 8;
    v4f x0 = *(const v4f*)(xrow + cb);
    v4f x1 = *(const v4f*)(xrow + cb + 4);
    v4f x2 = *(const v4f*)(xrow + cb + 16);
    v4f x3 = *(const v4f*)(xrow + cb + 20);
    BfPack a;
#pragma unroll
    for (int e = 0; e < 4; ++e) {
      a.us[e]      = f2bf(x0[e]);
      a.us[4 + e]  = f2bf(x1[e]);
      a.us[8 + e]  = f2bf(x2[e]);
      a.us[12 + e] = f2bf(x3[e]);
    }
    BfPack b;
    const v4f* bp = (const v4f*)(aPack + ((kt * 4 + nt) * 32 + lane) * 16);
    b.f4[0] = bp[0];
    b.f4[1] = bp[1];
    acc = __builtin_amdgcn_wmma_f32_16x16x32_bf16(
        false, a.bf, false, b.bf, (short)0, acc, false, false);
  }
  int n = nt * 16 + (lane & 15);
  int m0 = mb * 16 + half * 8;
#pragma unroll
  for (int r = 0; r < 8; ++r)
    h1[(m0 + r) * 64 + n] = f2bf(acc[r]);
}

// ---------------------------------------------------------------------------
// Kernel 4: out = h1 (4096x64 bf16) @ W1 -> f32 (4096x32768).
// One wave per (m_block, 16-tile N-chunk): 256 * 128 = 32768 waves.
// K = 64 -> exactly 2 WMMAs per 16x16 tile. A-operands loaded once per wave;
// W1 tiles stream from L2 (4 MB resident); output stored non-temporal.
// ---------------------------------------------------------------------------
__global__ __launch_bounds__(256) void gemm2(
    const unsigned short* __restrict__ h1,
    const unsigned short* __restrict__ wPack, float* __restrict__ out) {
  int wave = blockIdx.x * 8 + (threadIdx.x >> 5);  // 0..32767
  int lane = threadIdx.x & 31;
  int mb = wave >> 7;    // 0..255
  int nc = wave & 127;   // 0..127  (chunk of 16 N-tiles)
  int half = lane >> 4;

  const v4f* hp = (const v4f*)(h1 + (size_t)(mb * 16 + (lane & 15)) * 64);
  BfPack a0, a1;
  a0.f4[0] = hp[half];       // K =      half*8 .. +7
  a0.f4[1] = hp[2 + half];   // K = 16 + half*8 .. +7
  a1.f4[0] = hp[4 + half];   // K = 32 + half*8 .. +7
  a1.f4[1] = hp[6 + half];   // K = 48 + half*8 .. +7

  int m0 = mb * 16 + half * 8;
  for (int t = 0; t < 16; ++t) {
    int ntile = nc * 16 + t;
    const v4f* b0p = (const v4f*)(wPack + ((size_t)ntile * 32 + lane) * 16);
    const v4f* b1p =
        (const v4f*)(wPack + ((size_t)(2048 + ntile) * 32 + lane) * 16);
    BfPack b0, b1;
    b0.f4[0] = b0p[0]; b0.f4[1] = b0p[1];
    b1.f4[0] = b1p[0]; b1.f4[1] = b1p[1];
    v8f acc = {};
    acc = __builtin_amdgcn_wmma_f32_16x16x32_bf16(
        false, a0.bf, false, b0.bf, (short)0, acc, false, false);
    acc = __builtin_amdgcn_wmma_f32_16x16x32_bf16(
        false, a1.bf, false, b1.bf, (short)0, acc, false, false);
    float* op = out + (size_t)m0 * 32768 + (size_t)ntile * 16 + (lane & 15);
#pragma unroll
    for (int r = 0; r < 8; ++r)
      __builtin_nontemporal_store(acc[r], op + (size_t)r * 32768);
  }
}

extern "C" void kernel_launch(void* const* d_in, const int* in_sizes, int n_in,
                              void* d_out, int out_size, void* d_ws,
                              size_t ws_size, hipStream_t stream) {
  const float* x  = (const float*)d_in[0];
  const float* t0 = (const float*)d_in[1];
  const float* t1 = (const float*)d_in[2];
  const float* t2 = (const float*)d_in[3];
  const float* t3 = (const float*)d_in[4];
  const float* t4 = (const float*)d_in[5];
  const float* t5 = (const float*)d_in[6];
  float* out = (float*)d_out;

  unsigned short* aPack = (unsigned short*)d_ws;     // 65536  (128 KB)
  unsigned short* wPack = aPack + 65536;             // 2097152 (4 MB)
  unsigned short* h1    = wPack + 2097152;           // 262144 (512 KB)

  build_A_packed<<<256, 256, 0, stream>>>(t0, t1, t2, aPack);
  build_W1_packed<<<8192, 256, 0, stream>>>(t3, t4, t5, wPack);
  gemm1<<<128, 256, 0, stream>>>(x, aPack, h1);
  gemm2<<<4096, 256, 0, stream>>>(h1, wPack, out);
}
